// InterPointAttentionNet_45612552683807
// MI455X (gfx1250) — matrix operand order repaired and checked by reading, not compiled, not run on hardware
//
#include <hip/hip_runtime.h>
#include <hip/hip_bf16.h>
#include <math.h>

// Problem constants (match reference)
constexpr int Bn = 4, Nn = 256, Hn = 4, Cn = 64, Dn = 256, Fn = 1024, Pn = 128;
constexpr float LN_EPS = 1e-5f;

typedef __attribute__((ext_vector_type(16))) _Float16 v16h;
typedef __attribute__((ext_vector_type(8)))  float    v8f;

union AV { v16h v; _Float16 e[16]; };
union CV { v8f  v; float    e[8];  };

// ---------------------------------------------------------------------------
// Generic batched WMMA GEMM:  C = act( (A@B(^T) + bias + add) * alpha )
// One 16x16 output tile per wave32, K-loop in steps of 32 via
// v_wmma_f32_16x16x32_f16 (f32 accumulate). Batch dims: blockIdx.y ("h"),
// blockIdx.z ("b") with independent element strides per operand.
// TRANSB is a template parameter so the inner loop is branch-free.
//
// VGPR packing used here (wave32, contiguous k order):
//   A (16x32 f16): lane = M + 16*hi ; elem i -> k = hi*16 + i
//   B (32x16 f16): lane = N + 16*hi ; elem i -> k = hi*16 + i
//   C (16x16 f32): VGPR r -> row r + 8*hi, col = lane&15
// ---------------------------------------------------------------------------
template <int TRANSB>
__global__ __launch_bounds__(256) void gemm_wmma(
    const float* __restrict__ A, int lda, long sAh, long sAb,
    const float* __restrict__ Bm, int ldb, long sBh, long sBb,
    const float* __restrict__ bias,
    const float* __restrict__ add, int ldadd, long sDh, long sDb,
    float* __restrict__ C, int ldc, long sCh, long sCb,
    int M, int N, int Kdim, float alpha, int relu)
{
    const int hidx = blockIdx.y, bidx = blockIdx.z;
    A  += (size_t)hidx * sAh + (size_t)bidx * sAb;
    Bm += (size_t)hidx * sBh + (size_t)bidx * sBb;
    if (add) add += (size_t)hidx * sDh + (size_t)bidx * sDb;
    C  += (size_t)hidx * sCh + (size_t)bidx * sCb;

    const int tilesN = N >> 4;
    const int tiles  = (M >> 4) * tilesN;
    const int tile   = blockIdx.x * (blockDim.x >> 5) + (threadIdx.x >> 5);
    if (tile >= tiles) return;
    const int row0 = (tile / tilesN) << 4;
    const int col0 = (tile % tilesN) << 4;

    const int lane = threadIdx.x & 31;
    const int nl   = lane & 15;   // A row within tile / B & C column within tile
    const int hi   = lane >> 4;

    const float* __restrict__ arow = A + (size_t)(row0 + nl) * lda + hi * 16;
    const float* __restrict__ brow;
    if (TRANSB) {
        brow = Bm + (size_t)(col0 + nl) * ldb + hi * 16;     // contiguous 16 floats
    } else {
        brow = Bm + (size_t)(hi * 16) * ldb + (col0 + nl);   // column walk, stride ldb
    }

    CV acc;
#pragma unroll
    for (int r = 0; r < 8; ++r) acc.e[r] = 0.f;

    for (int k0 = 0; k0 < Kdim; k0 += 32) {
        // ---- A fragment: one contiguous 16-float run -> 4x b128 loads ----
        const float4 a0 = *(const float4*)(arow + k0);
        const float4 a1 = *(const float4*)(arow + k0 + 4);
        const float4 a2 = *(const float4*)(arow + k0 + 8);
        const float4 a3 = *(const float4*)(arow + k0 + 12);
        AV av;
        av.e[0]  = (_Float16)a0.x; av.e[1]  = (_Float16)a0.y;
        av.e[2]  = (_Float16)a0.z; av.e[3]  = (_Float16)a0.w;
        av.e[4]  = (_Float16)a1.x; av.e[5]  = (_Float16)a1.y;
        av.e[6]  = (_Float16)a1.z; av.e[7]  = (_Float16)a1.w;
        av.e[8]  = (_Float16)a2.x; av.e[9]  = (_Float16)a2.y;
        av.e[10] = (_Float16)a2.z; av.e[11] = (_Float16)a2.w;
        av.e[12] = (_Float16)a3.x; av.e[13] = (_Float16)a3.y;
        av.e[14] = (_Float16)a3.z; av.e[15] = (_Float16)a3.w;

        // ---- B fragment ----
        AV bv;
        if (TRANSB) {
            const float4 b0 = *(const float4*)(brow + k0);
            const float4 b1 = *(const float4*)(brow + k0 + 4);
            const float4 b2 = *(const float4*)(brow + k0 + 8);
            const float4 b3 = *(const float4*)(brow + k0 + 12);
            bv.e[0]  = (_Float16)b0.x; bv.e[1]  = (_Float16)b0.y;
            bv.e[2]  = (_Float16)b0.z; bv.e[3]  = (_Float16)b0.w;
            bv.e[4]  = (_Float16)b1.x; bv.e[5]  = (_Float16)b1.y;
            bv.e[6]  = (_Float16)b1.z; bv.e[7]  = (_Float16)b1.w;
            bv.e[8]  = (_Float16)b2.x; bv.e[9]  = (_Float16)b2.y;
            bv.e[10] = (_Float16)b2.z; bv.e[11] = (_Float16)b2.w;
            bv.e[12] = (_Float16)b3.x; bv.e[13] = (_Float16)b3.y;
            bv.e[14] = (_Float16)b3.z; bv.e[15] = (_Float16)b3.w;
        } else {
            const float* __restrict__ bp = brow + (size_t)k0 * ldb;
#pragma unroll
            for (int i = 0; i < 16; ++i)
                bv.e[i] = (_Float16)bp[(size_t)i * ldb];
        }

        acc.v = __builtin_amdgcn_wmma_f32_16x16x32_f16(
            false, av.v, false, bv.v, (short)0, acc.v, false, false);
    }

    const float bvv = bias ? bias[col0 + nl] : 0.f;
#pragma unroll
    for (int r = 0; r < 8; ++r) {
        const int row = row0 + r + hi * 8;
        float v = acc.e[r] + bvv;
        if (add) v += add[(size_t)row * ldadd + (col0 + nl)];
        v *= alpha;
        if (relu) v = fmaxf(v, 0.f);
        C[(size_t)row * ldc + (col0 + nl)] = v;
    }
}

// ---------------------------------------------------------------------------
// Fold channel-sum into pe2 weights: w2s[j][h] = sum_c pe2_w[j][h*C+c]
// b2s[h] = sum_c pe2_b[h*C+c]
// ---------------------------------------------------------------------------
__global__ void pe_reduce(const float* __restrict__ pe2_w,
                          const float* __restrict__ pe2_b,
                          float* __restrict__ w2s, float* __restrict__ b2s)
{
    const int j = threadIdx.x;               // 128 threads
    for (int h = 0; h < Hn; ++h) {
        float s = 0.f;
        for (int c = 0; c < Cn; ++c) s += pe2_w[j * (Hn * Cn) + h * Cn + c];
        w2s[j * Hn + h] = s;
    }
    if (j < Hn) {
        float s = 0.f;
        for (int c = 0; c < Cn; ++c) s += pe2_b[j * Cn + c];
        b2s[j] = s;
    }
}

// ---------------------------------------------------------------------------
// psum[b,h,m,k] = relu(loc[b,m,k,:] @ pe1_w + pe1_b) @ w2s[:,h] + b2s[h]
// One thread per (b,m,k) row; MLP weights staged in LDS.
// ---------------------------------------------------------------------------
__global__ __launch_bounds__(256) void pe_psum(
    const float* __restrict__ loc, const float* __restrict__ pe1_w,
    const float* __restrict__ pe1_b, const float* __restrict__ w2s,
    const float* __restrict__ b2s, float* __restrict__ psum)
{
    __shared__ float sw1[3 * Pn], sb1[Pn], sw2[Pn * Hn], sb2[Hn];
    const int tid = threadIdx.x;
    for (int i = tid; i < 3 * Pn; i += 256) sw1[i] = pe1_w[i];
    for (int i = tid; i < Pn;     i += 256) sb1[i] = pe1_b[i];
    for (int i = tid; i < Pn * Hn; i += 256) sw2[i] = w2s[i];
    if (tid < Hn) sb2[tid] = b2s[tid];
    __syncthreads();

    const size_t idx = (size_t)blockIdx.x * 256 + tid;   // over B*N*N
    if (idx >= (size_t)Bn * Nn * Nn) return;
    const float x = loc[idx * 3 + 0];
    const float y = loc[idx * 3 + 1];
    const float z = loc[idx * 3 + 2];
    float a0 = 0.f, a1 = 0.f, a2 = 0.f, a3 = 0.f;
#pragma unroll 4
    for (int j = 0; j < Pn; ++j) {
        float hval = fmaf(sw1[j], x, fmaf(sw1[Pn + j], y, fmaf(sw1[2 * Pn + j], z, sb1[j])));
        hval = fmaxf(hval, 0.f);
        a0 = fmaf(hval, sw2[j * Hn + 0], a0);
        a1 = fmaf(hval, sw2[j * Hn + 1], a1);
        a2 = fmaf(hval, sw2[j * Hn + 2], a2);
        a3 = fmaf(hval, sw2[j * Hn + 3], a3);
    }
    const size_t b   = idx >> 16;            // / (N*N)
    const size_t rem = idx & 65535;          // m*N + k
    float* out = psum + ((b * Hn) << 16) + rem;
    out[0u * 65536] = a0 + sb2[0];
    out[1u * 65536] = a1 + sb2[1];
    out[2u * 65536] = a2 + sb2[2];
    out[3u * 65536] = a3 + sb2[3];
}

// ---------------------------------------------------------------------------
// Row softmax over 256 columns, one wave32 per row (8 elems/lane).
// ---------------------------------------------------------------------------
__global__ __launch_bounds__(256) void softmax_rows(float* __restrict__ s, int rows)
{
    const int warp = threadIdx.x >> 5, lane = threadIdx.x & 31;
    const int row = blockIdx.x * 8 + warp;
    if (row >= rows) return;
    float* p = s + (size_t)row * Nn;
    float v[8];
    float mx = -1e30f;
#pragma unroll
    for (int i = 0; i < 8; ++i) { v[i] = p[lane + 32 * i]; mx = fmaxf(mx, v[i]); }
#pragma unroll
    for (int o = 16; o >= 1; o >>= 1) mx = fmaxf(mx, __shfl_xor(mx, o, 32));
    float sum = 0.f;
#pragma unroll
    for (int i = 0; i < 8; ++i) { v[i] = __expf(v[i] - mx); sum += v[i]; }
#pragma unroll
    for (int o = 16; o >= 1; o >>= 1) sum += __shfl_xor(sum, o, 32);
    const float inv = 1.f / sum;
#pragma unroll
    for (int i = 0; i < 8; ++i) p[lane + 32 * i] = v[i] * inv;
}

// ---------------------------------------------------------------------------
// LayerNorm over the whole [N, DIN] plane per batch (65536 elems),
// population variance, elementwise affine [N, DIN].
// ---------------------------------------------------------------------------
__global__ __launch_bounds__(1024) void ln2d(
    const float* __restrict__ x, const float* __restrict__ w,
    const float* __restrict__ bb, float* __restrict__ out)
{
    const int b = blockIdx.x;
    const int tid = threadIdx.x;
    const size_t base = (size_t)b * (Nn * Dn);
    float s = 0.f, ss = 0.f;
    for (int i = tid; i < Nn * Dn; i += 1024) {
        const float t = x[base + i];
        s += t; ss += t * t;
    }
#pragma unroll
    for (int o = 16; o >= 1; o >>= 1) { s += __shfl_xor(s, o, 32); ss += __shfl_xor(ss, o, 32); }
    __shared__ float ps[32], pss[32], stats[2];
    const int warp = tid >> 5, lane = tid & 31;
    if (lane == 0) { ps[warp] = s; pss[warp] = ss; }
    __syncthreads();
    if (warp == 0) {
        s = ps[lane]; ss = pss[lane];
#pragma unroll
        for (int o = 16; o >= 1; o >>= 1) { s += __shfl_xor(s, o, 32); ss += __shfl_xor(ss, o, 32); }
        if (lane == 0) {
            const float mean = s * (1.f / (Nn * Dn));
            const float var  = ss * (1.f / (Nn * Dn)) - mean * mean;
            stats[0] = mean;
            stats[1] = rsqrtf(var + LN_EPS);
        }
    }
    __syncthreads();
    const float mean = stats[0], rstd = stats[1];
    for (int i = tid; i < Nn * Dn; i += 1024) {
        const float t = (x[base + i] - mean) * rstd;
        out[base + i] = t * w[i] + bb[i];
    }
}

// ---------------------------------------------------------------------------
extern "C" void kernel_launch(void* const* d_in, const int* in_sizes, int n_in,
                              void* d_out, int out_size, void* d_ws, size_t ws_size,
                              hipStream_t stream)
{
    (void)in_sizes; (void)n_in; (void)out_size; (void)ws_size;
    const float* input_feat = (const float*)d_in[0];
    const float* loc   = (const float*)d_in[1];
    const float* Kw    = (const float*)d_in[2];
    const float* Kb    = (const float*)d_in[3];
    const float* Qw    = (const float*)d_in[4];
    const float* Qb    = (const float*)d_in[5];
    const float* Vw    = (const float*)d_in[6];
    const float* Vb    = (const float*)d_in[7];
    const float* Fw    = (const float*)d_in[8];
    const float* Fb    = (const float*)d_in[9];
    const float* pe1w  = (const float*)d_in[10];
    const float* pe1b  = (const float*)d_in[11];
    const float* pe2w  = (const float*)d_in[12];
    const float* pe2b  = (const float*)d_in[13];
    const float* f1w   = (const float*)d_in[14];
    const float* f1b   = (const float*)d_in[15];
    const float* f2w   = (const float*)d_in[16];
    const float* f2b   = (const float*)d_in[17];
    const float* ln1w  = (const float*)d_in[18];
    const float* ln1b  = (const float*)d_in[19];
    const float* ln2w  = (const float*)d_in[20];
    const float* ln2b  = (const float*)d_in[21];
    float* out = (float*)d_out;

    // Workspace layout (floats), heavily aliased:
    float* ws   = (float*)d_ws;
    float* w2s  = ws;                       //   512 : folded pe2 weights
    float* b2s  = ws + 512;                 //     4 : folded pe2 bias
    float* big  = ws + 1024;                // 1048576: psum -> scores -> attn -> ffn hidden
    float* kbuf = big + 1048576;            // 262144 : K -> merged
    float* qbuf = kbuf + 262144;            // 262144 : Q -> res1 -> res2
    float* vbuf = qbuf + 262144;            // 262144 : V -> out1

    const long NN  = (long)Nn * Nn;         // 65536
    const long HNN = (long)Hn * NN;

    auto launch_gemm = [&](const float* A, int lda, long sAh, long sAb,
                           const float* Bm, int ldb, long sBh, long sBb,
                           const float* bias,
                           const float* add, int ldadd, long sDh, long sDb,
                           float* C, int ldc, long sCh, long sCb,
                           int M, int N, int Kd, float alpha, int transB, int relu,
                           int gy, int gz) {
        const int tiles = (M / 16) * (N / 16);
        dim3 grid((tiles + 7) / 8, gy, gz);
        if (transB)
            gemm_wmma<1><<<grid, 256, 0, stream>>>(A, lda, sAh, sAb, Bm, ldb, sBh, sBb,
                                                   bias, add, ldadd, sDh, sDb,
                                                   C, ldc, sCh, sCb, M, N, Kd, alpha, relu);
        else
            gemm_wmma<0><<<grid, 256, 0, stream>>>(A, lda, sAh, sAb, Bm, ldb, sBh, sBb,
                                                   bias, add, ldadd, sDh, sDb,
                                                   C, ldc, sCh, sCb, M, N, Kd, alpha, relu);
    };

    // 1) Fold pe2 over channels; 2) psum[b,h,m,k] via tiny MLP
    pe_reduce<<<1, 128, 0, stream>>>(pe2w, pe2b, w2s, b2s);
    pe_psum<<<(Bn * Nn * Nn) / 256, 256, 0, stream>>>(loc, pe1w, pe1b, w2s, b2s, big);

    // 3) QKV projections: [1024,256] @ [256,256] + bias
    launch_gemm(input_feat, Dn, 0, 0, Kw, Hn * Cn, 0, 0, Kb,
                nullptr, 0, 0, 0, kbuf, Hn * Cn, 0, 0,
                Bn * Nn, Hn * Cn, Dn, 1.f, 0, 0, 1, 1);
    launch_gemm(input_feat, Dn, 0, 0, Qw, Hn * Cn, 0, 0, Qb,
                nullptr, 0, 0, 0, qbuf, Hn * Cn, 0, 0,
                Bn * Nn, Hn * Cn, Dn, 1.f, 0, 0, 1, 1);
    launch_gemm(input_feat, Dn, 0, 0, Vw, Hn * Cn, 0, 0, Vb,
                nullptr, 0, 0, 0, vbuf, Hn * Cn, 0, 0,
                Bn * Nn, Hn * Cn, Dn, 1.f, 0, 0, 1, 1);

    // 4) scores[b,h] = (K @ Q^T + psum) / sqrt(C), in-place add into `big`
    launch_gemm(kbuf, Dn, Cn, (long)Nn * Dn,
                qbuf, Dn, Cn, (long)Nn * Dn,     // transB: rows of Q are columns
                nullptr,
                big, Nn, NN, HNN,
                big, Nn, NN, HNN,
                Nn, Nn, Cn, 0.125f, 1, 0, Hn, Bn);

    // 5) softmax over k (last dim), in place
    softmax_rows<<<(Bn * Hn * Nn + 7) / 8, 256, 0, stream>>>(big, Bn * Hn * Nn);

    // 6) merged[b,j,h*C+c] = attn[b,h] @ V[b,:,h*C..] -> kbuf
    launch_gemm(big, Nn, NN, HNN,
                vbuf, Dn, Cn, (long)Nn * Dn,
                nullptr, nullptr, 0, 0, 0,
                kbuf, Dn, Cn, (long)Nn * Dn,
                Nn, Cn, Nn, 1.f, 0, 0, Hn, Bn);

    // 7) output projection + residual -> qbuf (pre-LN1)
    launch_gemm(kbuf, Dn, 0, 0, Fw, Dn, 0, 0, Fb,
                input_feat, Dn, 0, 0,
                qbuf, Dn, 0, 0,
                Bn * Nn, Dn, Dn, 1.f, 0, 0, 1, 1);

    // 8) LN1 over [N, DIN] per batch -> vbuf (= out1)
    ln2d<<<Bn, 1024, 0, stream>>>(qbuf, ln1w, ln1b, vbuf);

    // 9) FFN1 with ReLU -> big  [1024,1024]
    launch_gemm(vbuf, Dn, 0, 0, f1w, Fn, 0, 0, f1b,
                nullptr, 0, 0, 0,
                big, Fn, 0, 0,
                Bn * Nn, Fn, Dn, 1.f, 0, 1, 1, 1);

    // 10) FFN2 + residual(out1) -> qbuf (pre-LN2)
    launch_gemm(big, Fn, 0, 0, f2w, Dn, 0, 0, f2b,
                vbuf, Dn, 0, 0,
                qbuf, Dn, 0, 0,
                Bn * Nn, Dn, Fn, 1.f, 0, 0, 1, 1);

    // 11) LN2 -> final output
    ln2d<<<Bn, 1024, 0, stream>>>(qbuf, ln2w, ln2b, out);
}
